// AttentionTE_15221364097676
// MI455X (gfx1250) — matrix operand order repaired and checked by reading, not compiled, not run on hardware
//
#include <hip/hip_runtime.h>
#include <hip/hip_bf16.h>
#include <math.h>

// ---------------------------------------------------------------------------
// CDNA5 (gfx1250) fused pair-bias gated attention, bf16 WMMA throughout.
// Roofline: 512MB fp32 bias read dominates (~22us @ 23.3TB/s). The bias tile
// is streamed with GLOBAL_LOAD_ASYNC_TO_LDS_B128 (ASYNCcnt) overlapped with
// the QK^T WMMAs; all GEMMs run on v_wmma_f32_16x16x32_bf16 (fp32 accum).
// ---------------------------------------------------------------------------

typedef __attribute__((ext_vector_type(16))) __bf16 v16bf;
typedef __attribute__((ext_vector_type(8)))  __bf16 v8bf;
typedef __attribute__((ext_vector_type(8)))  float  v8f;

constexpr int kDim  = 1024;
constexpr int kH    = 16;
constexpr int kC    = 64;
constexpr int kB    = 2;
constexpr int kN    = 2048;
constexpr int kRows = kB * kN;          // 4096
constexpr float kEps    = 1e-5f;
constexpr float kQScale = 0.125f;       // 64^-0.5

// ---- WMMA fragment loaders (wave32; A and B both use contiguous k = 16h + s, not the ISA table order) ----------
// 16-bit A 16x32: lanes 0-15 hold M=0..15 with K {0..15}; lanes 16-31
// hold K {16..31}.  Two contiguous 16B loads per lane.
__device__ __forceinline__ v16bf load_a_frag(const __bf16* base, int ld, int k0, int lane) {
  const int m    = lane & 15;
  const int koff = (lane >> 4) * 16;
  const __bf16* p = base + (size_t)m * ld + k0 + koff;
  v8bf lo = *reinterpret_cast<const v8bf*>(p);
  v8bf hi = *reinterpret_cast<const v8bf*>(p + 8);
  v16bf a;
#pragma unroll
  for (int i = 0; i < 8; ++i) { a[i] = lo[i]; a[i + 8] = hi[i]; }
  return a;
}

// 16-bit B 32x16: lane holds column n=l&15, 16 consecutive K values for its
// half (l<16 -> K0..15, l>=16 -> K16..31).  W is row-major [n][k] so this is a
// single contiguous 32B load (contraction axis is contiguous for all GEMMs).
__device__ __forceinline__ v16bf load_b_frag(const __bf16* w, int ld, int n0, int k0, int lane) {
  const int n  = lane & 15;
  const int kb = (lane >> 4) * 16;
  return *reinterpret_cast<const v16bf*>(w + (size_t)(n0 + n) * ld + k0 + kb);
}

__device__ __forceinline__ v8f wmma_bf16(v16bf a, v16bf b, v8f c) {
  return __builtin_amdgcn_wmma_f32_16x16x32_bf16(false, a, false, b, (short)0, c, false, false);
}

// Async DMA: one 16B chunk global -> LDS, tracked by ASYNCcnt (ISA 10.x / 15.18.3).
// lds_off is the LDS byte offset (= low 32 bits of a generic shared pointer,
// per ISA 10.2 aperture rules).
__device__ __forceinline__ void async_ld_b128(unsigned lds_off, const void* gaddr) {
  asm volatile("global_load_async_to_lds_b128 %0, %1, off"
               :: "v"(lds_off), "v"(gaddr) : "memory");
}
__device__ __forceinline__ void wait_async0() {
  asm volatile("s_wait_asynccnt 0x0" ::: "memory");
}

// One wave computes a 16x64 fp32 tile of A(16xK) * W^T (W row-major [n][k]).
__device__ __forceinline__ void wave_gemm_16x64(const __bf16* A, int lda,
                                                const __bf16* W, int ldw, int n0,
                                                int K, int lane, v8f acc[4]) {
  for (int k0 = 0; k0 < K; k0 += 32) {
    v16bf a = load_a_frag(A, lda, k0, lane);
#pragma unroll
    for (int t = 0; t < 4; ++t) {
      v16bf b = load_b_frag(W, ldw, n0 + t * 16, k0, lane);
      acc[t] = wmma_bf16(a, b, acc[t]);
    }
  }
}

// ---------------------------------------------------------------------------
__global__ void cvt_bf16_kernel(const float* __restrict__ src, __bf16* __restrict__ dst, int n) {
  int i = blockIdx.x * blockDim.x + threadIdx.x;
  const int stride = gridDim.x * blockDim.x;
  for (; i < n; i += stride) dst[i] = (__bf16)src[i];
}

// LayerNorm: one 256-thread block per row of x (B*N rows, DIM=1024).
__global__ void __launch_bounds__(256) ln_kernel(const float* __restrict__ x,
                                                 const float* __restrict__ w,
                                                 const float* __restrict__ b,
                                                 __bf16* __restrict__ xn) {
  const int row = blockIdx.x;
  const int tid = threadIdx.x;
  const float* xr = x + (size_t)row * kDim;
  float v[4];
  float s = 0.f, s2 = 0.f;
#pragma unroll
  for (int i = 0; i < 4; ++i) {
    float f = xr[tid + i * 256];
    v[i] = f; s += f; s2 += f * f;
  }
  __shared__ float r1[256], r2[256];
  r1[tid] = s; r2[tid] = s2;
  __syncthreads();
  for (int off = 128; off > 0; off >>= 1) {
    if (tid < off) { r1[tid] += r1[tid + off]; r2[tid] += r2[tid + off]; }
    __syncthreads();
  }
  const float mu  = r1[0] * (1.f / kDim);
  const float var = r2[0] * (1.f / kDim) - mu * mu;
  const float inv = rsqrtf(var + kEps);
#pragma unroll
  for (int i = 0; i < 4; ++i) {
    const int d = tid + i * 256;
    xn[(size_t)row * kDim + d] = (__bf16)((v[i] - mu) * inv * w[d] + b[d]);
  }
}

// QKV projection: t = xn @ w_qkv^T, scattered into q/k/v as (B,H,N,C) bf16,
// with q pre-scaled by C^-0.5.
__global__ void __launch_bounds__(128) qkv_kernel(const __bf16* __restrict__ xn,
                                                  const __bf16* __restrict__ wqkv,
                                                  __bf16* __restrict__ qo,
                                                  __bf16* __restrict__ ko,
                                                  __bf16* __restrict__ vo) {
  const int lane = threadIdx.x & 31;
  const int wv   = threadIdx.x >> 5;
  const int m0   = blockIdx.x * 64 + wv * 16;
  const int n0   = blockIdx.y * 64;
  v8f acc[4] = {};
  wave_gemm_16x64(xn + (size_t)m0 * kDim, kDim, wqkv, kDim, n0, kDim, lane, acc);
  const int col = lane & 15;
  const int rh  = (lane >> 4) * 8;
#pragma unroll
  for (int t = 0; t < 4; ++t) {
    const int e  = n0 + t * 16 + col;
    const int h  = e / (3 * kC);
    const int r3 = e % (3 * kC);
    const int which = r3 / kC;
    const int c     = r3 % kC;
    __bf16* dst = (which == 0) ? qo : ((which == 1) ? ko : vo);
    const float sc = (which == 0) ? kQScale : 1.0f;
#pragma unroll
    for (int r = 0; r < 8; ++r) {
      const int row = m0 + rh + r;
      const int bb = row / kN, n = row % kN;
      dst[(((size_t)bb * kH + h) * kN + n) * kC + c] = (__bf16)(acc[t][r] * sc);
    }
  }
}

// Flash attention over 64-query tiles; 4 waves x 16 queries each. Online
// softmax. Bias tile async-DMA'd to LDS (overlapped with QK^T WMMAs); P staged
// via LDS (A-layout re-entry); V staged transposed via LDS (B-layout re-entry).
__global__ void __launch_bounds__(128) attn_kernel(const __bf16* __restrict__ q,
                                                   const __bf16* __restrict__ kmat,
                                                   const __bf16* __restrict__ vmat,
                                                   const float* __restrict__ bias,
                                                   const int* __restrict__ mask,
                                                   __bf16* __restrict__ ybuf) {
  const int qt = blockIdx.x;       // query tile (N/64)
  const int h  = blockIdx.y;
  const int b  = blockIdx.z;
  const int lane = threadIdx.x & 31;
  const int wv   = threadIdx.x >> 5;
  const int col  = lane & 15;
  const int rh   = (lane >> 4) * 8;
  const size_t bh = (size_t)b * kH + h;

  const int qb0 = qt * 64;             // block's first query row
  const int q0  = qb0 + wv * 16;       // wave's first query row
  const __bf16* Q  = q    + (bh * kN + q0) * kC;
  const __bf16* Km = kmat + bh * kN * kC;
  const __bf16* Vm = vmat + bh * kN * kC;
  const float*  Bs = bias + bh * (size_t)kN * kN;
  const int*    Mk = mask + b * kN;

  __shared__ __bf16 Pbuf[4 * 16 * 64];   // per-wave P tile (A layout source)
  __shared__ __bf16 Vt[64 * 64];         // V chunk transposed: Vt[c][key]
  __shared__ float  Btile[64 * 64];      // bias tile: Btile[qrow][key], 16KB
  __bf16* Pw = Pbuf + wv * (16 * 64);
  const unsigned btile_base = (unsigned)(size_t)&Btile[0];

  float mrow[8], lsum[8];
  v8f yacc[4] = {};
#pragma unroll
  for (int r = 0; r < 8; ++r) { mrow[r] = -1e30f; lsum[r] = 0.f; }

  for (int kb = 0; kb < kN; kb += 64) {
    __syncthreads();   // previous iteration's Vt/Btile reads complete
    // --- async DMA bias tile (64 rows x 256B) into LDS: 1024 x 16B chunks ---
#pragma unroll
    for (int i = 0; i < 8; ++i) {
      const int id  = threadIdx.x + i * 128;
      const int row = id >> 4;             // 0..63 query row in tile
      const int cx  = id & 15;             // 16B chunk within row
      const float* g = Bs + (size_t)(qb0 + row) * kN + kb + cx * 4;
      async_ld_b128(btile_base + (unsigned)(row * 256 + cx * 16), g);
    }
    // --- cooperative transpose-stage of V chunk into LDS ---
    for (int idx = threadIdx.x; idx < 64 * 64; idx += 128) {
      const int key = idx >> 6, c = idx & 63;
      Vt[c * 64 + key] = Vm[(size_t)(kb + key) * kC + c];
    }
    // --- S = Q K^T : 16 queries x 64 keys, contraction over C=64 ---
    v8f s[4] = {};
#pragma unroll
    for (int k0 = 0; k0 < kC; k0 += 32) {
      v16bf a = load_a_frag(Q, kC, k0, lane);
#pragma unroll
      for (int t = 0; t < 4; ++t) {
        v16bf bf = load_b_frag(Km, kC, kb + t * 16, k0, lane);
        s[t] = wmma_bf16(a, bf, s[t]);
      }
    }
    wait_async0();     // my async chunks of Btile landed
    __syncthreads();   // everyone's Btile chunks + Vt visible
    // --- + bias (from LDS), + mask (-inf surrogate) ---
#pragma unroll
    for (int t = 0; t < 4; ++t) {
      const int key = kb + t * 16 + col;
      const float mval = (Mk[key] == 0) ? -1e30f : 0.f;
#pragma unroll
      for (int r = 0; r < 8; ++r) {
        const int lrow = wv * 16 + rh + r;          // row within 64-query tile
        s[t][r] = s[t][r] + Btile[lrow * 64 + t * 16 + col] + mval;
      }
    }
    // --- online softmax: row stats live across 16-lane halves (C/D layout) ---
    float mnew[8], scale[8], rs[8];
#pragma unroll
    for (int r = 0; r < 8; ++r) {
      float mx = fmaxf(fmaxf(s[0][r], s[1][r]), fmaxf(s[2][r], s[3][r]));
#pragma unroll
      for (int off = 1; off < 16; off <<= 1) mx = fmaxf(mx, __shfl_xor(mx, off));
      mnew[r]  = fmaxf(mrow[r], mx);
      scale[r] = __expf(mrow[r] - mnew[r]);
      rs[r]    = 0.f;
    }
#pragma unroll
    for (int t = 0; t < 4; ++t)
#pragma unroll
      for (int r = 0; r < 8; ++r) {
        const float p = __expf(s[t][r] - mnew[r]);
        rs[r] += p;
        Pw[(rh + r) * 64 + t * 16 + col] = (__bf16)p;   // ds_store_b16
      }
#pragma unroll
    for (int r = 0; r < 8; ++r) {
#pragma unroll
      for (int off = 1; off < 16; off <<= 1) rs[r] += __shfl_xor(rs[r], off);
      lsum[r] = lsum[r] * scale[r] + rs[r];
      mrow[r] = mnew[r];
    }
#pragma unroll
    for (int t = 0; t < 4; ++t)
#pragma unroll
      for (int r = 0; r < 8; ++r) yacc[t][r] *= scale[r];

    // --- Y += P V  (A from LDS Pw, B from LDS Vt; Pw is same-wave) ---
#pragma unroll
    for (int k0 = 0; k0 < 64; k0 += 32) {
      v16bf pa = load_a_frag(Pw, 64, k0, lane);
#pragma unroll
      for (int t = 0; t < 4; ++t) {
        v16bf vb = load_b_frag(Vt, 64, t * 16, k0, lane);
        yacc[t] = wmma_bf16(pa, vb, yacc[t]);
      }
    }
  }
  // normalize and write ybuf as (B,N,DIM) bf16, d = h*C + c
#pragma unroll
  for (int r = 0; r < 8; ++r) {
    const float inv = 1.f / fmaxf(lsum[r], 1e-20f);
    const int n = q0 + rh + r;
#pragma unroll
    for (int t = 0; t < 4; ++t) {
      const int d = h * kC + t * 16 + col;
      ybuf[((size_t)b * kN + n) * kDim + d] = (__bf16)(yacc[t][r] * inv);
    }
  }
}

// Gate: g = sigmoid(xn @ w_g^T + b_g); ybar = g * y
__global__ void __launch_bounds__(128) gate_kernel(const __bf16* __restrict__ xn,
                                                   const __bf16* __restrict__ wg,
                                                   const float* __restrict__ bg,
                                                   const __bf16* __restrict__ ybuf,
                                                   __bf16* __restrict__ ybar) {
  const int lane = threadIdx.x & 31;
  const int wv   = threadIdx.x >> 5;
  const int m0   = blockIdx.x * 64 + wv * 16;
  const int n0   = blockIdx.y * 64;
  v8f acc[4] = {};
  wave_gemm_16x64(xn + (size_t)m0 * kDim, kDim, wg, kDim, n0, kDim, lane, acc);
  const int col = lane & 15, rh = (lane >> 4) * 8;
#pragma unroll
  for (int t = 0; t < 4; ++t) {
    const int e = n0 + t * 16 + col;
    const float be = bg[e];
#pragma unroll
    for (int r = 0; r < 8; ++r) {
      const size_t row = m0 + rh + r;
      const float g = 1.f / (1.f + __expf(-(acc[t][r] + be)));
      ybar[row * kDim + e] = (__bf16)(g * (float)ybuf[row * kDim + e]);
    }
  }
}

// Output projection: out = ybar @ w_o^T + b_o (fp32 output)
__global__ void __launch_bounds__(128) out_kernel(const __bf16* __restrict__ ybar,
                                                  const __bf16* __restrict__ wo,
                                                  const float* __restrict__ bo,
                                                  float* __restrict__ out) {
  const int lane = threadIdx.x & 31;
  const int wv   = threadIdx.x >> 5;
  const int m0   = blockIdx.x * 64 + wv * 16;
  const int n0   = blockIdx.y * 64;
  v8f acc[4] = {};
  wave_gemm_16x64(ybar + (size_t)m0 * kDim, kDim, wo, kDim, n0, kDim, lane, acc);
  const int col = lane & 15, rh = (lane >> 4) * 8;
#pragma unroll
  for (int t = 0; t < 4; ++t) {
    const int e = n0 + t * 16 + col;
    const float be = bo[e];
#pragma unroll
    for (int r = 0; r < 8; ++r) {
      const size_t row = m0 + rh + r;
      out[row * kDim + e] = acc[t][r] + be;
    }
  }
}

// ---------------------------------------------------------------------------
extern "C" void kernel_launch(void* const* d_in, const int* in_sizes, int n_in,
                              void* d_out, int out_size, void* d_ws, size_t ws_size,
                              hipStream_t stream) {
  (void)in_sizes; (void)n_in; (void)out_size; (void)ws_size;
  const float* x    = (const float*)d_in[0];
  const float* bias = (const float*)d_in[1];
  const int*   mask = (const int*)d_in[2];
  const float* lnw  = (const float*)d_in[3];
  const float* lnb  = (const float*)d_in[4];
  const float* wqkv = (const float*)d_in[5];
  const float* wo   = (const float*)d_in[6];
  const float* bo   = (const float*)d_in[7];
  const float* wg   = (const float*)d_in[8];
  const float* bg   = (const float*)d_in[9];
  float* out = (float*)d_out;

  char* p = (char*)d_ws;
  auto take = [&](size_t bytes) { char* r = p; p += (bytes + 255) & ~size_t(255); return r; };
  __bf16* xn    = (__bf16*)take((size_t)kRows * kDim * 2);      // 8 MB
  __bf16* wqkvb = (__bf16*)take((size_t)3 * kDim * kDim * 2);   // 6 MB
  __bf16* wgb   = (__bf16*)take((size_t)kDim * kDim * 2);       // 2 MB
  __bf16* wob   = (__bf16*)take((size_t)kDim * kDim * 2);       // 2 MB
  __bf16* qb    = (__bf16*)take((size_t)kRows * kDim * 2);      // 8 MB
  __bf16* kb    = (__bf16*)take((size_t)kRows * kDim * 2);      // 8 MB
  __bf16* vb    = (__bf16*)take((size_t)kRows * kDim * 2);      // 8 MB
  __bf16* ybuf  = (__bf16*)take((size_t)kRows * kDim * 2);      // 8 MB
  __bf16* ybar  = (__bf16*)take((size_t)kRows * kDim * 2);      // 8 MB

  cvt_bf16_kernel<<<512, 256, 0, stream>>>(wqkv, wqkvb, 3 * kDim * kDim);
  cvt_bf16_kernel<<<256, 256, 0, stream>>>(wg, wgb, kDim * kDim);
  cvt_bf16_kernel<<<256, 256, 0, stream>>>(wo, wob, kDim * kDim);
  ln_kernel<<<kRows, 256, 0, stream>>>(x, lnw, lnb, xn);
  qkv_kernel<<<dim3(kRows / 64, (3 * kDim) / 64), 128, 0, stream>>>(xn, wqkvb, qb, kb, vb);
  attn_kernel<<<dim3(kN / 64, kH, kB), 128, 0, stream>>>(qb, kb, vb, bias, mask, ybuf);
  gate_kernel<<<dim3(kRows / 64, kDim / 64), 128, 0, stream>>>(xn, wgb, bg, ybuf, ybar);
  out_kernel<<<dim3(kRows / 64, kDim / 64), 128, 0, stream>>>(ybar, wob, bo, out);
}
